// AttentiveFP_78005196030502
// MI455X (gfx1250) — compile-verified, matrix operand order checked by reading, not run on hardware
//
#include <hip/hip_runtime.h>
#include <hip/hip_bf16.h>
#include <math.h>

typedef __attribute__((ext_vector_type(16))) __bf16 bf16x16;
typedef __attribute__((ext_vector_type(8)))  float  v8f;

static constexpr int cN   = 100000;
static constexpr int cE   = 300000;
static constexpr int cG   = 4096;
static constexpr int cHID = 200;
static constexpr int cH   = 8;
static constexpr int cL   = 4;
static constexpr int cT   = 5;

// ---------------------------------------------------------------- utilities
__global__ void fill_kernel(float* p, float v, int n) {
  int i = blockIdx.x * blockDim.x + threadIdx.x;
  if (i < n) p[i] = v;
}

__device__ __forceinline__ void atomicMaxF(float* addr, float val) {
  int* ai = (int*)addr;
  int old = __float_as_int(*addr);
  while (__int_as_float(old) < val) {
    int assumed = old;
    old = atomicCAS(ai, assumed, __float_as_int(val));
    if (old == assumed) break;
  }
}

// ------------------------------------------------- pack B into WMMA fragment order
// fragment id = (kc*NT + t)*32 + lane ; 16 bf16 elems contiguous per lane.
// lane: col = t*16 + (lane&15), khalf = lane>>4
// elem j in [0,8): k = kc*32 + khalf*8 + j ; elem 8+j: k = kc*32 + 16 + khalf*8 + j
__global__ void pack_b_kernel(const float* __restrict__ B, __bf16* __restrict__ Bp,
                              int K, int Nn) {
  int NT = (Nn + 15) >> 4;
  int KC = (K + 31) >> 5;
  int total = KC * NT * 32 * 16;
  int idx = blockIdx.x * blockDim.x + threadIdx.x;
  if (idx >= total) return;
  int e    = idx & 15;
  int lane = (idx >> 4) & 31;
  int ft   = idx >> 9;          // kc*NT + t
  int t    = ft % NT;
  int kc   = ft / NT;
  int col  = t * 16 + (lane & 15);
  int k    = kc * 32 + ((e >> 3) ? 16 : 0) + ((lane >> 4) << 3) + (e & 7);
  float v = (k < K && col < Nn) ? B[(size_t)k * Nn + col] : 0.f;
  Bp[idx] = (__bf16)v;
}

// ------------------------------------------------- bf16 WMMA GEMM: C = A*B (+bias)(+C)
// grid.x = M/16 (M must be multiple of 16), block = 128 (4 waves), wave w -> col tiles [4w,4w+4)
__global__ __launch_bounds__(128)
void gemm_bf16_kernel(const float* __restrict__ A, const __bf16* __restrict__ Bp,
                      const float* __restrict__ bias, float* __restrict__ C,
                      int M, int K, int Nn, int beta) {
  int lane  = threadIdx.x & 31;
  int wave  = __builtin_amdgcn_readfirstlane((int)threadIdx.x >> 5);  // scalar
  int mtile = blockIdx.x;
  int NT = (Nn + 15) >> 4;
  int KC = (K + 31) >> 5;
  int row = mtile * 16 + (lane & 15);
  int kh  = (lane >> 4) << 3;   // 0 or 8
  const float* Arow = A + (size_t)row * K;
  int t0 = wave * 4;            // scalar
  int nt = NT - t0;             // scalar: valid tiles for this wave

  v8f zero = {0.f, 0.f, 0.f, 0.f, 0.f, 0.f, 0.f, 0.f};
  v8f acc0 = zero, acc1 = zero, acc2 = zero, acc3 = zero;

  for (int kc = 0; kc < KC; ++kc) {
    int kb = kc << 5;
    bf16x16 fa;
    if (kb + 32 <= K) {
      // full chunk: 4 aligned b128 loads (kb+kh is a multiple of 8 floats)
      float4 a0 = *(const float4*)(Arow + kb + kh);
      float4 a1 = *(const float4*)(Arow + kb + kh + 4);
      float4 a2 = *(const float4*)(Arow + kb + 16 + kh);
      float4 a3 = *(const float4*)(Arow + kb + 16 + kh + 4);
      fa[0]  = (__bf16)a0.x; fa[1]  = (__bf16)a0.y; fa[2]  = (__bf16)a0.z; fa[3]  = (__bf16)a0.w;
      fa[4]  = (__bf16)a1.x; fa[5]  = (__bf16)a1.y; fa[6]  = (__bf16)a1.z; fa[7]  = (__bf16)a1.w;
      fa[8]  = (__bf16)a2.x; fa[9]  = (__bf16)a2.y; fa[10] = (__bf16)a2.z; fa[11] = (__bf16)a2.w;
      fa[12] = (__bf16)a3.x; fa[13] = (__bf16)a3.y; fa[14] = (__bf16)a3.z; fa[15] = (__bf16)a3.w;
    } else {
      // tail chunk (runs at most once): per-element guards
#pragma unroll
      for (int j = 0; j < 8; ++j) {
        int k0 = kb + kh + j;
        int k1 = kb + 16 + kh + j;
        fa[j]     = (k0 < K) ? (__bf16)Arow[k0] : (__bf16)0.f;
        fa[j + 8] = (k1 < K) ? (__bf16)Arow[k1] : (__bf16)0.f;
      }
    }
    const __bf16* base = Bp + (((size_t)kc * NT + t0) * 32 + lane) * 16;
    // scalar branches: EXEC stays all-ones across every WMMA
    if (nt > 0)
      acc0 = __builtin_amdgcn_wmma_f32_16x16x32_bf16(
          false, fa, false, *(const bf16x16*)(base), (short)0, acc0, false, false);
    if (nt > 1)
      acc1 = __builtin_amdgcn_wmma_f32_16x16x32_bf16(
          false, fa, false, *(const bf16x16*)(base + 512), (short)0, acc1, false, false);
    if (nt > 2)
      acc2 = __builtin_amdgcn_wmma_f32_16x16x32_bf16(
          false, fa, false, *(const bf16x16*)(base + 1024), (short)0, acc2, false, false);
    if (nt > 3)
      acc3 = __builtin_amdgcn_wmma_f32_16x16x32_bf16(
          false, fa, false, *(const bf16x16*)(base + 1536), (short)0, acc3, false, false);
  }

  int colL  = lane & 15;
  int rofs  = (lane >> 4) << 3;
  int mbase = mtile * 16 + rofs;
#define STORE_TILE(ACC, TI)                                    \
  {                                                            \
    int col = (TI) * 16 + colL;                                \
    if (col < Nn) {                                            \
      float bv = bias ? bias[col] : 0.f;                       \
      _Pragma("unroll")                                        \
      for (int r = 0; r < 8; ++r) {                            \
        size_t off = (size_t)(mbase + r) * Nn + col;           \
        float v = (ACC)[r] + bv;                               \
        if (beta) v += C[off];                                 \
        C[off] = v;                                            \
      }                                                        \
    }                                                          \
  }
  if (nt > 0) STORE_TILE(acc0, t0);
  if (nt > 1) STORE_TILE(acc1, t0 + 1);
  if (nt > 2) STORE_TILE(acc2, t0 + 2);
  if (nt > 3) STORE_TILE(acc3, t0 + 3);
#undef STORE_TILE
}

// ------------------------------------------------- degree + segment-sum of edge_attr
__global__ void deg_s6_kernel(const float* __restrict__ ea, const int* __restrict__ dst,
                              float* deg, float* s6, int E) {
  int e = blockIdx.x * blockDim.x + threadIdx.x;
  if (e >= E) return;
  int d = dst[e];
  atomicAdd(&deg[d], 1.f);
#pragma unroll
  for (int j = 0; j < 6; ++j) atomicAdd(&s6[d * 6 + j], ea[e * 6 + j]);
}

// ------------------------------------------------- fold attention vectors into [200,8] mats
__global__ void prep_att1_kernel(const float* __restrict__ linW, const float* __restrict__ linEW,
                                 const float* __restrict__ attS, const float* __restrict__ attD,
                                 const float* __restrict__ attE,
                                 float* w_src, float* w_dst, float* w_edge) {
  int idx = blockIdx.x * blockDim.x + threadIdx.x;
  if (idx >= 3 * cHID * cH) return;
  int mat = idx / (cHID * cH);
  int r   = idx % (cHID * cH);
  int k   = r >> 3, hh = r & 7;
  const float* W   = (mat == 2) ? linEW : linW;
  const float* att = (mat == 0) ? attS : (mat == 1) ? attD : attE;
  float acc = 0.f;
#pragma unroll
  for (int c = 0; c < 25; ++c) acc += W[k * cHID + hh * 25 + c] * att[hh * 25 + c];
  float* out = (mat == 0) ? w_src : (mat == 1) ? w_dst : w_edge;
  out[k * 8 + hh] = acc;
}

__global__ void prep_att2_kernel(const float* __restrict__ eW, const float* __restrict__ eB,
                                 const float* __restrict__ w_edge, float* we6, float* be) {
  int idx = threadIdx.x;
  if (idx < 48) {
    int j = idx >> 3, hh = idx & 7;
    float acc = 0.f;
    for (int k = 0; k < cHID; ++k) acc += eW[j * cHID + k] * w_edge[k * 8 + hh];
    we6[idx] = acc;
  } else if (idx < 56) {
    int hh = idx - 48;
    float acc = 0.f;
    for (int k = 0; k < cHID; ++k) acc += eB[k] * w_edge[k * 8 + hh];
    be[hh] = acc;
  }
}

// ------------------------------------------------- per-node a_src/a_dst + folded self-loop a_e
__global__ void node_att_kernel(const float* __restrict__ h, const float* __restrict__ w_src,
                                const float* __restrict__ w_dst, const float* __restrict__ s6,
                                const float* __restrict__ deg, const float* __restrict__ we6,
                                const float* __restrict__ be,
                                float* a_src, float* a_dst, float* aeL, int n) {
  int idx = blockIdx.x * blockDim.x + threadIdx.x;
  if (idx >= n * 8) return;
  int node = idx >> 3, hh = idx & 7;
  const float* hr = h + (size_t)node * cHID;
  float as = 0.f, ad = 0.f;
  for (int k = 0; k < cHID; ++k) {
    float hv = hr[k];
    as += hv * w_src[k * 8 + hh];
    ad += hv * w_dst[k * 8 + hh];
  }
  a_src[idx] = as;
  a_dst[idx] = ad;
  float d = deg[node];
  float inv = 1.f / fmaxf(d, 1.f);
  float ae = d * be[hh];
#pragma unroll
  for (int j = 0; j < 6; ++j) ae += s6[node * 6 + j] * we6[j * 8 + hh];
  aeL[idx] = ae * inv;
}

// ------------------------------------------------- folded real-edge a_e
__global__ void edge_ae_kernel(const float* __restrict__ ea, const float* __restrict__ we6,
                               const float* __restrict__ be, float* a_e, int E) {
  int idx = blockIdx.x * blockDim.x + threadIdx.x;
  if (idx >= E * 8) return;
  int e = idx >> 3, hh = idx & 7;
  float acc = be[hh];
  const float* row = ea + (size_t)e * 6;
#pragma unroll
  for (int j = 0; j < 6; ++j) acc += row[j] * we6[j * 8 + hh];
  a_e[idx] = acc;
}

// ------------------------------------------------- logits + segment max
__global__ void logit_max_kernel(const float* __restrict__ a_src, const float* __restrict__ a_dst,
                                 const float* __restrict__ a_e, const int* __restrict__ src,
                                 const int* __restrict__ dst, float* alpha, float* m, int E) {
  int idx = blockIdx.x * blockDim.x + threadIdx.x;
  if (idx >= E * 8) return;
  int e = idx >> 3, hh = idx & 7;
  int sI = src[e], dI = dst[e];
  float v = a_src[sI * 8 + hh] + a_dst[dI * 8 + hh] + a_e[idx];
  v = v > 0.f ? v : 0.2f * v;
  alpha[idx] = v;
  atomicMaxF(&m[dI * 8 + hh], v);
}

__global__ void loop_logit_max_kernel(const float* __restrict__ a_src,
                                      const float* __restrict__ a_dst,
                                      const float* __restrict__ aeL,
                                      float* lalpha, float* m, int n) {
  int idx = blockIdx.x * blockDim.x + threadIdx.x;
  if (idx >= n * 8) return;
  float v = a_src[idx] + a_dst[idx] + aeL[idx];
  v = v > 0.f ? v : 0.2f * v;
  lalpha[idx] = v;
  atomicMaxF(&m[idx], v);
}

// ------------------------------------------------- exp + segment sum
__global__ void exp_sum_kernel(float* alpha, const float* __restrict__ m, float* s,
                               const int* __restrict__ dst, int E) {
  int idx = blockIdx.x * blockDim.x + threadIdx.x;
  if (idx >= E * 8) return;
  int e = idx >> 3, hh = idx & 7;
  int dI = dst[e];
  float ex = __expf(alpha[idx] - m[dI * 8 + hh]);
  alpha[idx] = ex;
  atomicAdd(&s[dI * 8 + hh], ex);
}

__global__ void loop_exp_sum_kernel(float* lalpha, const float* __restrict__ m, float* s, int n) {
  int idx = blockIdx.x * blockDim.x + threadIdx.x;
  if (idx >= n * 8) return;
  float ex = __expf(lalpha[idx] - m[idx]);
  lalpha[idx] = ex;
  atomicAdd(&s[idx], ex);
}

__global__ void inv_s_kernel(float* s, int n) {
  int idx = blockIdx.x * blockDim.x + threadIdx.x;
  if (idx < n) s[idx] = 1.f / (s[idx] + 1e-16f);
}

// ------------------------------------------------- h_new = gat_bias then scatter messages
__global__ void init_hnew_kernel(float* hnew, const float* __restrict__ bias, int n) {
  int idx = blockIdx.x * blockDim.x + threadIdx.x;
  if (idx >= n * cHID) return;
  hnew[idx] = bias[idx % cHID];
}

// one wave (32 lanes) per (edge or self-loop); lanes stride the 200 feature dims
__global__ void aggregate_kernel(const float* __restrict__ xs, const float* __restrict__ alpha,
                                 const float* __restrict__ lalpha, const float* __restrict__ sinv,
                                 const int* __restrict__ src, const int* __restrict__ dst,
                                 float* hnew, int E, int n) {
  int idx  = blockIdx.x * blockDim.x + threadIdx.x;
  // lanes of one wave share idx>>5 (256-thread blocks are wave-aligned) -> scalarize
  int edge = __builtin_amdgcn_readfirstlane(idx >> 5);
  int lane = idx & 31;
  if (edge >= E + n) return;
  int sI, dI;
  const float* al;
  if (edge < E) {
    sI = src[edge]; dI = dst[edge];
    al = alpha + (size_t)edge * 8;
  } else {
    sI = dI = edge - E;
    al = lalpha + (size_t)(edge - E) * 8;
  }
  const float* xr = xs + (size_t)sI * cHID;
  const float* sv = sinv + (size_t)dI * 8;
  float* hr = hnew + (size_t)dI * cHID;
  for (int d = lane; d < cHID; d += 32) {
    int hh = d / 25;
    atomicAdd(&hr[d], xr[d] * (al[hh] * sv[hh]));
  }
}

// ------------------------------------------------- global_add_pool
__global__ void pool_kernel(const float* __restrict__ hnew, const int* __restrict__ batch,
                            float* pooled, int n) {
  int idx = blockIdx.x * blockDim.x + threadIdx.x;
  if (idx >= n * cHID) return;
  int node = idx / cHID, d = idx % cHID;
  atomicAdd(&pooled[(size_t)batch[node] * cHID + d], hnew[idx]);
}

// ------------------------------------------------- task heads
__global__ __launch_bounds__(128)
void heads_kernel(const float* __restrict__ readouts, const float* __restrict__ w1,
                  const float* __restrict__ b1, const float* __restrict__ w2,
                  const float* __restrict__ b2, float* out, int G) {
  __shared__ float r[cHID];
  __shared__ float hid[100];
  int g = blockIdx.x;
  for (int i = threadIdx.x; i < cHID; i += blockDim.x) r[i] = readouts[(size_t)g * cHID + i];
  __syncthreads();
  const float SIGv[5] = {1.f, 0.f, 0.f, 1.f, 1.f};
  for (int t = 0; t < cT; ++t) {
    for (int j = threadIdx.x; j < 100; j += blockDim.x) {
      float acc = b1[t * 100 + j];
      const float* w = w1 + (size_t)t * cHID * 100 + j;
      for (int k = 0; k < cHID; ++k) acc += r[k] * w[k * 100];
      hid[j] = fmaxf(acc, 0.f);
    }
    __syncthreads();
    if (threadIdx.x == 0) {
      float z = b2[t];
      for (int j = 0; j < 100; ++j) z += hid[j] * w2[t * 100 + j];
      out[(size_t)t * G + g] = (SIGv[t] > 0.f) ? 1.f / (1.f + __expf(-z)) : z;
    }
    __syncthreads();
  }
}

// ================================================================= launcher
static inline void fill_async(float* p, float v, int n, hipStream_t st) {
  fill_kernel<<<(n + 255) / 256, 256, 0, st>>>(p, v, n);
}

extern "C" void kernel_launch(void* const* d_in, const int* in_sizes, int n_in,
                              void* d_out, int out_size, void* d_ws, size_t ws_size,
                              hipStream_t stream) {
  (void)in_sizes; (void)n_in; (void)out_size; (void)ws_size;
  const float* x          = (const float*)d_in[0];
  const float* edge_attr  = (const float*)d_in[1];
  const float* node_emb_w = (const float*)d_in[2];
  const float* node_emb_b = (const float*)d_in[3];
  const float* edge_emb_w = (const float*)d_in[4];
  const float* edge_emb_b = (const float*)d_in[5];
  const float* lin_w      = (const float*)d_in[6];
  const float* lin_edge_w = (const float*)d_in[7];
  const float* att_src    = (const float*)d_in[8];
  const float* att_dst    = (const float*)d_in[9];
  const float* att_edge   = (const float*)d_in[10];
  const float* gat_bias   = (const float*)d_in[11];
  const float* readout_w  = (const float*)d_in[12];
  const float* readout_b  = (const float*)d_in[13];
  const float* head_w1    = (const float*)d_in[14];
  const float* head_b1    = (const float*)d_in[15];
  const float* head_w2    = (const float*)d_in[16];
  const float* head_b2    = (const float*)d_in[17];
  const int*   edge_index = (const int*)d_in[18];
  const int*   batch      = (const int*)d_in[19];
  const int* srcI = edge_index;
  const int* dstI = edge_index + cE;
  float* out = (float*)d_out;

  // ---- workspace carving (all float-aligned; Bpack ends up 32B aligned)
  float* ws = (float*)d_ws;
  float* h        = ws;                    ws += (size_t)cN * cHID;
  float* hnew     = ws;                    ws += (size_t)cN * cHID;
  float* xs       = ws;                    ws += (size_t)cN * cHID;
  float* a_src    = ws;                    ws += (size_t)cN * 8;
  float* a_dst    = ws;                    ws += (size_t)cN * 8;
  float* aeL      = ws;                    ws += (size_t)cN * 8;
  float* mbuf     = ws;                    ws += (size_t)cN * 8;
  float* sbuf     = ws;                    ws += (size_t)cN * 8;
  float* lalpha   = ws;                    ws += (size_t)cN * 8;
  float* deg      = ws;                    ws += (size_t)cN;
  float* s6       = ws;                    ws += (size_t)cN * 6;
  float* a_e      = ws;                    ws += (size_t)cE * 8;
  float* alpha    = ws;                    ws += (size_t)cE * 8;
  float* pooled   = ws;                    ws += (size_t)cG * cHID;
  float* readouts = ws;                    ws += (size_t)cG * cHID;
  float* w_src    = ws;                    ws += cHID * 8;
  float* w_dst    = ws;                    ws += cHID * 8;
  float* w_edge   = ws;                    ws += cHID * 8;
  float* we6      = ws;                    ws += 48;
  float* be       = ws;                    ws += 8;   // total preceding floats % 8 == 0
  __bf16* Bpack   = (__bf16*)ws;                      // 7*13*512 = 46592 bf16 max

  const int packTotal = 7 * 13 * 512;   // KCmax * NTmax * 32 * 16 elems

  // ---- one-time prep
  fill_async(deg, 0.f, cN, stream);
  fill_async(s6, 0.f, cN * 6, stream);
  fill_async(readouts, 0.f, cG * cHID, stream);
  deg_s6_kernel<<<(cE + 255) / 256, 256, 0, stream>>>(edge_attr, dstI, deg, s6, cE);

  // h = x @ node_emb_w + b   (M=100000, K=64, Nn=200)
  pack_b_kernel<<<(2 * 13 * 512 + 255) / 256, 256, 0, stream>>>(node_emb_w, Bpack, 64, cHID);
  gemm_bf16_kernel<<<cN / 16, 128, 0, stream>>>(x, Bpack, node_emb_b, h, cN, 64, cHID, 0);

  float* hcur = h;
  float* hnx  = hnew;

  for (int l = 0; l < cL; ++l) {
    const float* linW  = lin_w      + (size_t)l * cHID * cHID;
    const float* linEW = lin_edge_w + (size_t)l * cHID * cHID;
    const float* attS  = att_src  + (size_t)l * cH * 25;
    const float* attD  = att_dst  + (size_t)l * cH * 25;
    const float* attE  = att_edge + (size_t)l * cH * 25;

    prep_att1_kernel<<<(3 * cHID * 8 + 255) / 256, 256, 0, stream>>>(
        linW, linEW, attS, attD, attE, w_src, w_dst, w_edge);
    prep_att2_kernel<<<1, 64, 0, stream>>>(edge_emb_w, edge_emb_b, w_edge, we6, be);

    // xs = h @ lin_w[l]
    pack_b_kernel<<<(packTotal + 255) / 256, 256, 0, stream>>>(linW, Bpack, cHID, cHID);
    gemm_bf16_kernel<<<cN / 16, 128, 0, stream>>>(hcur, Bpack, nullptr, xs, cN, cHID, cHID, 0);

    node_att_kernel<<<(cN * 8 + 255) / 256, 256, 0, stream>>>(
        hcur, w_src, w_dst, s6, deg, we6, be, a_src, a_dst, aeL, cN);
    edge_ae_kernel<<<(cE * 8 + 255) / 256, 256, 0, stream>>>(edge_attr, we6, be, a_e, cE);

    fill_async(mbuf, -3.0e38f, cN * 8, stream);
    fill_async(sbuf, 0.f, cN * 8, stream);
    logit_max_kernel<<<(cE * 8 + 255) / 256, 256, 0, stream>>>(
        a_src, a_dst, a_e, srcI, dstI, alpha, mbuf, cE);
    loop_logit_max_kernel<<<(cN * 8 + 255) / 256, 256, 0, stream>>>(
        a_src, a_dst, aeL, lalpha, mbuf, cN);
    exp_sum_kernel<<<(cE * 8 + 255) / 256, 256, 0, stream>>>(alpha, mbuf, sbuf, dstI, cE);
    loop_exp_sum_kernel<<<(cN * 8 + 255) / 256, 256, 0, stream>>>(lalpha, mbuf, sbuf, cN);
    inv_s_kernel<<<(cN * 8 + 255) / 256, 256, 0, stream>>>(sbuf, cN * 8);

    init_hnew_kernel<<<(cN * cHID + 255) / 256, 256, 0, stream>>>(hnx, gat_bias + (size_t)l * cHID, cN);
    {
      long long work = (long long)(cE + cN) * 32;
      aggregate_kernel<<<(int)((work + 255) / 256), 256, 0, stream>>>(
          xs, alpha, lalpha, sbuf, srcI, dstI, hnx, cE, cN);
    }

    fill_async(pooled, 0.f, cG * cHID, stream);
    pool_kernel<<<(cN * cHID + 255) / 256, 256, 0, stream>>>(hnx, batch, pooled, cN);

    // readouts += pooled @ readout_w[l] + readout_b[l]
    pack_b_kernel<<<(packTotal + 255) / 256, 256, 0, stream>>>(
        readout_w + (size_t)l * cHID * cHID, Bpack, cHID, cHID);
    gemm_bf16_kernel<<<cG / 16, 128, 0, stream>>>(
        pooled, Bpack, readout_b + (size_t)l * cHID, readouts, cG, cHID, cHID, 1);

    float* tmp = hcur; hcur = hnx; hnx = tmp;
  }

  heads_kernel<<<cG, 128, 0, stream>>>(readouts, head_w1, head_b1, head_w2, head_b2, out, cG);
}
